// MyEdgeConvBlock_13477607375096
// MI455X (gfx1250) — matrix operand order repaired and checked by reading, not compiled, not run on hardware
//
#include <hip/hip_runtime.h>
#include <hip/hip_bf16.h>

typedef __attribute__((ext_vector_type(16))) _Float16 v16h;
typedef __attribute__((ext_vector_type(8)))  _Float16 v8h;
typedef __attribute__((ext_vector_type(8)))  float    v8f;

#define N_NODES   10000
#define E_EDGES   640000
#define C_IN      128
#define C_OUT     128
#define H_DIM     256
#define BN_EPS    1e-5f

#define THREADS          512                 // 16 wave32 = 2 tile groups x 8 waves
#define TILE_STEPS       16                  // iterations per block
#define TILES_PER_BLOCK  (2 * TILE_STEPS)    // 32 edge tiles of 16 edges
#define NUM_BLOCKS       (E_EDGES / (TILES_PER_BLOCK * 16))   // 1250

// LDS layout (halves unless noted):
//   w1s  : 16nt * 8ks * 32lane * 16e = 65536 h = 131072 B
//   w2s  :  8nt * 8ks * 32lane * 16e = 32768 h =  65536 B
//   feat : 4 buffers (2 groups x double-buffer) * 4096 h = 16384 h = 32768 B
//   hbuf : 2 groups * 4096 h = 8192 h = 16384 B
//   dstIdx : 4 * 16 int = 256 B
#define FEAT_HALFS 4096
#define SMEM_BYTES (131072 + 65536 + 32768 + 16384 + 256)     // 246016 B < 320KB/WGP

// ---------- A-fragment swizzle (ISA 7.12.2, 16-bit A 16x32) ----------
// lane = hh*16 + m ; element e ; K = ks*32 + (e<8 ? e+8*hh : e+8+8*hh)
// Inverse: (m, K) -> flat half index into [ks][lane][e].
// For K in an aligned group of 8 (K0 = 8*q), all 8 halves are contiguous.
__device__ __forceinline__ int aswz(int m, int K) {
  int ks = K >> 5;
  int kk = K & 31;
  int hh = (kk >> 3) & 1;
  int e  = kk - 8 * (kk >> 4) - 8 * hh;
  return ((ks * 32 + hh * 16 + m) << 4) + e;
}

// ---------- prep kernels ----------
__global__ void bn_prep_kernel(const float* __restrict__ gamma, const float* __restrict__ beta,
                               const float* __restrict__ mean,  const float* __restrict__ var,
                               float* __restrict__ scale, float* __restrict__ shift) {
  int c = threadIdx.x;
  if (c < C_IN) {
    float sc = gamma[c] * rsqrtf(var[c] + BN_EPS);
    scale[c] = sc;
    shift[c] = beta[c] - mean[c] * sc;
  }
}

// Pre-swizzle weights into the B-fragment register layout:
// flat = ((nt*8 + ks)*32 + lane)*16 + e ; K = ks*32 + e + 8*(lane>>4) + 8*(e>>3) ; N = nt*16 + (lane&15)
__global__ void w_prep_kernel(const float* __restrict__ W1, const float* __restrict__ W2,
                              _Float16* __restrict__ w1p, _Float16* __restrict__ w2p) {
  int i = blockIdx.x * blockDim.x + threadIdx.x;
  if (i < 65536) {
    int e = i & 15, lane = (i >> 4) & 31, ks = (i >> 9) & 7, nt = i >> 12;
    int K = ks * 32 + e + 8 * (lane >> 4) + 8 * (e >> 3);
    int N = nt * 16 + (lane & 15);
    w1p[i] = (_Float16)W1[K * H_DIM + N];
  } else if (i < 65536 + 32768) {
    int l = i - 65536;
    int e = l & 15, lane = (l >> 4) & 31, ks = (l >> 9) & 7, nt = l >> 12;
    int K = ks * 32 + e + 8 * (lane >> 4) + 8 * (e >> 3);
    int N = nt * 16 + (lane & 15);
    w2p[l] = (_Float16)W2[K * C_OUT + N];
  }
}

__global__ void init_out_kernel(float* __restrict__ out, int n) {
  int i = blockIdx.x * blockDim.x + threadIdx.x;
  if (i < n) out[i] = 0.0f;
}

// ---------- staging helpers ----------
struct Stage {
  float4 d0, d1, s0, s1;
  int d;
};

__device__ __forceinline__ void stage_load(const float* __restrict__ x,
                                           const int* __restrict__ ei,
                                           int e0, int m, int cg, Stage& st) {
  int e = e0 + m;
  int s = ei[e];               // src j
  st.d  = ei[E_EDGES + e];     // dst i
  const float4* xd4 = (const float4*)(x + (long)st.d * C_IN + cg);
  const float4* xs4 = (const float4*)(x + (long)s  * C_IN + cg);
  st.d0 = xd4[0]; st.d1 = xd4[1]; st.s0 = xs4[0]; st.s1 = xs4[1];
}

__device__ __forceinline__ void stage_commit(const Stage& st, _Float16* __restrict__ featBuf,
                                             int* __restrict__ dstBuf,
                                             const float* sc, const float* sh,
                                             int m, int cg, bool writeDst) {
  v8h xi, xd;
#pragma unroll
  for (int j = 0; j < 8; ++j) {
    float xdv = (j < 4) ? (&st.d0.x)[j] : (&st.d1.x)[j - 4];
    float xsv = (j < 4) ? (&st.s0.x)[j] : (&st.s1.x)[j - 4];
    xi[j] = (_Float16)(xdv * sc[j] + sh[j]);        // xn_i
    xd[j] = (_Float16)((xsv - xdv) * sc[j]);        // xn_j - xn_i (shift cancels)
  }
  // contiguous 16B stores in A-swizzled layout (single ds_store_b128 each)
  *(v8h*)(featBuf + aswz(m, cg))        = xi;
  *(v8h*)(featBuf + aswz(m, C_IN + cg)) = xd;
  if (writeDst) dstBuf[m] = st.d;
}

// ---------- main fused kernel ----------
__global__ void __launch_bounds__(THREADS)
edgeconv_main_kernel(const float* __restrict__ x, const int* __restrict__ ei,
                     const float* __restrict__ b1g, const float* __restrict__ b2g,
                     const float* __restrict__ scale_g, const float* __restrict__ shift_g,
                     const _Float16* __restrict__ w1g, const _Float16* __restrict__ w2g,
                     float* __restrict__ out) {
  extern __shared__ char smem_raw[];
  _Float16* w1s  = (_Float16*)smem_raw;            // 65536 halves
  _Float16* w2s  = w1s + 65536;                    // 32768 halves
  _Float16* feat = w2s + 32768;                    // 4 * 4096 halves (A-swizzled)
  _Float16* hbuf = feat + 4 * FEAT_HALFS;          // 2 * 4096 halves (A-swizzled)
  int*      dstIdx = (int*)(hbuf + 2 * FEAT_HALFS);// 4 * 16 ints

  const int tid  = threadIdx.x;
  const int wave = tid >> 5;
  const int g    = wave >> 3;       // compute tile-group 0/1
  const int sw   = wave & 7;        // wave within group
  const int lane = tid & 31;
  const int nl   = lane & 15;
  const int hh   = lane >> 4;

  // staging role: 16 threads per edge, 8 channels each, per group
  const int g_s  = tid >> 8;        // staging tile-group 0/1
  const int m_s  = (tid >> 4) & 15; // edge within tile
  const int cg_s = (tid & 15) << 3; // channel group
  const bool wDst = (tid & 15) == 0;

  // per-thread BN constants (channel group is fixed per thread)
  float sc[8], sh[8];
#pragma unroll
  for (int j = 0; j < 8; ++j) { sc[j] = scale_g[cg_s + j]; sh[j] = shift_g[cg_s + j]; }

  // ---- cooperatively load swizzled weights into LDS ----
  {
    const uint4* srcv = (const uint4*)w1g;         // w1g,w2g contiguous in ws
    uint4*       dstv = (uint4*)w1s;
    const int n16 = (131072 + 65536) / 16;         // 12288 uint4
#pragma unroll 2
    for (int i = tid; i < n16; i += THREADS) dstv[i] = srcv[i];
  }

  const v16h* w1v = (const v16h*)w1s;
  const v16h* w2v = (const v16h*)w2s;
  const float bias1a = b1g[sw * 16 + nl];
  const float bias1b = b1g[(sw + 8) * 16 + nl];
  const float bias2  = b2g[sw * 16 + nl];

  const int tileBase = blockIdx.x * TILES_PER_BLOCK;
  Stage st;

  // prologue: stage tile pair 0 into buffer 0
  stage_load(x, ei, (tileBase + 0 * 2 + g_s) * 16, m_s, cg_s, st);
  stage_commit(st, feat + (g_s * 2 + 0) * FEAT_HALFS, dstIdx + (g_s * 2 + 0) * 16,
               sc, sh, m_s, cg_s, wDst);
  __syncthreads();

  for (int t = 0; t < TILE_STEPS; ++t) {
    const int cur = t & 1, nxt = cur ^ 1;
    const bool more = (t + 1 < TILE_STEPS);

    // issue next tile's gather early: latency overlaps layer-1 WMMA chain
    if (more) stage_load(x, ei, (tileBase + (t + 1) * 2 + g_s) * 16, m_s, cg_s, st);

    // ---- layer 1: h = ReLU(feat @ W1 + b1); wave -> N-tiles {sw, sw+8} ----
    {
      const v16h* fv = (const v16h*)(feat + (g * 2 + cur) * FEAT_HALFS);
      _Float16*   hb = hbuf + g * FEAT_HALFS;
      v8f acc0 = {}, acc1 = {};
#pragma unroll
      for (int ks = 0; ks < 8; ++ks) {
        v16h a  = fv[ks * 32 + lane];
        v16h b0 = w1v[(sw * 8 + ks) * 32 + lane];
        acc0 = __builtin_amdgcn_wmma_f32_16x16x32_f16(false, a, false, b0, (short)0, acc0, false, false);
        v16h b1f = w1v[((sw + 8) * 8 + ks) * 32 + lane];
        acc1 = __builtin_amdgcn_wmma_f32_16x16x32_f16(false, a, false, b1f, (short)0, acc1, false, false);
      }
#pragma unroll
      for (int r = 0; r < 8; ++r) {
        int M = r + 8 * hh;   // C-layout: VGPR r, lane half hh
        hb[aswz(M, sw * 16 + nl)]       = (_Float16)fmaxf(acc0[r] + bias1a, 0.0f);
        hb[aswz(M, (sw + 8) * 16 + nl)] = (_Float16)fmaxf(acc1[r] + bias1b, 0.0f);
      }
    }

    // commit next tile's feat into the alternate buffer (waits its loads here)
    if (more) stage_commit(st, feat + (g_s * 2 + nxt) * FEAT_HALFS,
                           dstIdx + (g_s * 2 + nxt) * 16, sc, sh, m_s, cg_s, wDst);
    __syncthreads();   // hbuf ready for L2; feat[nxt] ready for next iter

    // ---- layer 2: msg = h @ W2 + b2 ; fused segment-max via atomic max ----
    {
      const v16h* hv = (const v16h*)(hbuf + g * FEAT_HALFS);
      const int*  di = dstIdx + (g * 2 + cur) * 16;
      v8f acc = {};
#pragma unroll
      for (int ks = 0; ks < 8; ++ks) {
        v16h a = hv[ks * 32 + lane];
        v16h b = w2v[(sw * 8 + ks) * 32 + lane];
        acc = __builtin_amdgcn_wmma_f32_16x16x32_f16(false, a, false, b, (short)0, acc, false, false);
      }
#pragma unroll
      for (int r = 0; r < 8; ++r) {
        int M = r + 8 * hh;
        float msg = acc[r] + bias2;
        // Final relu(max(...)) with 0-init output: only positive msgs matter;
        // positive-float bit patterns order like ints -> integer atomic max.
        if (msg > 0.0f) {
          atomicMax((int*)out + (long)di[M] * C_OUT + sw * 16 + nl,
                    __float_as_int(msg));
        }
      }
    }
    __syncthreads();   // hbuf reusable
  }
}

extern "C" void kernel_launch(void* const* d_in, const int* in_sizes, int n_in,
                              void* d_out, int out_size, void* d_ws, size_t ws_size,
                              hipStream_t stream) {
  const float* x     = (const float*)d_in[0];
  const int*   ei    = (const int*)d_in[1];
  const float* gamma = (const float*)d_in[2];
  const float* beta  = (const float*)d_in[3];
  const float* mean  = (const float*)d_in[4];
  const float* var   = (const float*)d_in[5];
  const float* W1    = (const float*)d_in[6];
  const float* b1    = (const float*)d_in[7];
  const float* W2    = (const float*)d_in[8];
  const float* b2    = (const float*)d_in[9];
  float* out = (float*)d_out;

  // workspace layout: scale(128f) shift(128f) | w1p(65536 h) w2p(32768 h) ~198KB
  char* ws = (char*)d_ws;
  float*    scale = (float*)ws;
  float*    shift = scale + C_IN;
  _Float16* w1p   = (_Float16*)(ws + 1024);
  _Float16* w2p   = w1p + 65536;

  bn_prep_kernel<<<1, 128, 0, stream>>>(gamma, beta, mean, var, scale, shift);
  w_prep_kernel<<<(65536 + 32768) / 256, 256, 0, stream>>>(W1, W2, w1p, w2p);
  init_out_kernel<<<(N_NODES * C_OUT + 255) / 256, 256, 0, stream>>>(out, N_NODES * C_OUT);

  edgeconv_main_kernel<<<NUM_BLOCKS, THREADS, SMEM_BYTES, stream>>>(
      x, ei, b1, b2, scale, shift, w1p, w2p, out);
}
